// ArcfaceCombLoss_53738630808193
// MI455X (gfx1250) — compile-verified, matrix operand order checked by reading, not run on hardware
//
#include <hip/hip_runtime.h>
#include <hip/hip_bf16.h>
#include <math.h>

typedef __bf16 bf16;
typedef __attribute__((ext_vector_type(16))) __bf16 v16bf;
typedef __attribute__((ext_vector_type(8)))  __bf16 v8bf;
typedef __attribute__((ext_vector_type(8)))  float   v8f;

namespace {
constexpr int NB   = 2048;    // batch
constexpr int NC   = 20000;   // classes
constexpr int ND   = 512;     // dim
constexpr int LMAXC= 8;
constexpr int BM   = 32;      // rows per block (2 x 16 row tiles)
constexpr int BN   = 256;     // cols per block (8 waves x 2 tiles x 16)
constexpr int BK   = 64;      // K chunk staged in LDS
constexpr int NCB  = (NC + BN - 1) / BN;  // 79 column blocks
constexpr int LDP  = BK + 8;  // padded LDS row stride (bf16 elems, 16B-aligned)

constexpr float S_SCALE = 30.0f;
constexpr float COS_M   = 0.87758256189037276f;   // cos(0.5)
constexpr float SIN_M   = 0.47942553860420301f;   // sin(0.5)
constexpr float THR     = -0.87758256189037276f;  // cos(pi-0.5)
constexpr float MMV     = 0.23971276930210151f;   // sin(pi-0.5)*0.5
}

// ---------------------------------------------------------------------------
// Row L2 norms for both w (rows 0..NC-1) and f (rows NC..NC+NB-1).
// ---------------------------------------------------------------------------
__global__ __launch_bounds__(256) void norms_kernel(const float* __restrict__ f,
                                                    const float* __restrict__ w,
                                                    float* __restrict__ wn,
                                                    float* __restrict__ fn) {
  const int lane = threadIdx.x & 31;
  const int wave = threadIdx.x >> 5;
  const int row  = blockIdx.x * 8 + wave;
  const float* src;
  float* dst;
  if (row < NC)            { src = w + (size_t)row * ND;        dst = wn + row; }
  else if (row < NC + NB)  { src = f + (size_t)(row - NC) * ND; dst = fn + (row - NC); }
  else return;
  float s = 0.f;
#pragma unroll
  for (int j = 0; j < 4; ++j) {
    float4 v = *(const float4*)(src + j * 128 + lane * 4);
    s += v.x * v.x + v.y * v.y + v.z * v.z + v.w * v.w;
  }
#pragma unroll
  for (int d = 1; d < 32; d <<= 1) s += __shfl_xor(s, d, 32);
  if (lane == 0) *dst = sqrtf(s);
}

// ---------------------------------------------------------------------------
// WMMA helpers (16-bit A/B 16x32 layout, ISA 7.12.2).
// ---------------------------------------------------------------------------
__device__ __forceinline__ v16bf ldfrag(const bf16* __restrict__ rowp, int kb) {
  v8bf p0 = *(const v8bf*)(rowp + kb);
  v8bf p1 = *(const v8bf*)(rowp + kb + 16);
  return __builtin_shufflevector(p0, p1, 0, 1, 2, 3, 4, 5, 6, 7,
                                 8, 9, 10, 11, 12, 13, 14, 15);
}

__device__ __forceinline__ v8f wmma3(v8f acc, v16bf ah, v16bf al,
                                     v16bf bh, v16bf bl) {
  // split-bf16 fp32 emulation: hi*hi + hi*lo + lo*hi
  acc = __builtin_amdgcn_wmma_f32_16x16x32_bf16(false, ah, false, bh, (short)0, acc, false, false);
  acc = __builtin_amdgcn_wmma_f32_16x16x32_bf16(false, ah, false, bl, (short)0, acc, false, false);
  acc = __builtin_amdgcn_wmma_f32_16x16x32_bf16(false, al, false, bh, (short)0, acc, false, false);
  return acc;
}

__device__ __forceinline__ void store_hilo(bf16* hi, bf16* lo, float x) {
  bf16 h = (bf16)x;
  *hi = h;
  *lo = (bf16)(x - (float)h);
}

__device__ __forceinline__ float arcface_out(float dot, float fnv, float wnv,
                                             float lab, bool valid) {
  float cosv = dot / fmaxf(fnv * wnv, 1e-8f);
  float sine = sqrtf(fminf(fmaxf(1.f - cosv * cosv, 0.f), 1.f));
  float phi  = cosv * COS_M - sine * SIN_M;
  phi = (cosv > THR) ? phi : (cosv - MMV);
  return valid ? S_SCALE * (lab * phi + (1.f - lab) * cosv) : -INFINITY;
}

// ---------------------------------------------------------------------------
// Fused GEMM (split-bf16 WMMA) + ArcFace elementwise + per-tile softmax stats.
// Block = 256 threads (8 waves). Tile = 32 rows x 256 cols, K=512 in BK chunks.
// ---------------------------------------------------------------------------
__global__ __launch_bounds__(256) void gemm_stats_kernel(
    const float* __restrict__ f, const float* __restrict__ labels,
    const float* __restrict__ w, const float* __restrict__ fn,
    const float* __restrict__ wn, float* __restrict__ stats) {
  __shared__ __align__(16) bf16 aHi[BM][LDP];
  __shared__ __align__(16) bf16 aLo[BM][LDP];
  __shared__ __align__(16) bf16 bHi[BN][LDP];
  __shared__ __align__(16) bf16 bLo[BN][LDP];
  __shared__ float sRed[8][BM];
  __shared__ float rowMax[BM];

  const int tid  = threadIdx.x;
  const int wave = tid >> 5;
  const int lane = tid & 31;
  const int rowBase = blockIdx.x * BM;
  const int colBase = blockIdx.y * BN;

  v8f acc00 = {}, acc01 = {};  // rows 0-15 x two col tiles
  v8f acc10 = {}, acc11 = {};  // rows 16-31 x two col tiles

  const int m     = lane & 15;
  const int khalf = lane >> 4;

  for (int k0 = 0; k0 < ND; k0 += BK) {
    // ---- batched global loads (unconditional, clamped addresses) ----
    float4 av[2];
#pragma unroll
    for (int j = 0; j < 2; ++j) {
      const int e = (j * 256 + tid) * 4;
      av[j] = *(const float4*)(&f[(size_t)(rowBase + (e >> 6)) * ND + k0 + (e & 63)]);
    }
    float4 bv[16];
#pragma unroll
    for (int it = 0; it < 16; ++it) {
      const int e   = (it * 256 + tid) * 4;
      const int gcl = min(colBase + (e >> 6), NC - 1);
      bv[it] = *(const float4*)(&w[(size_t)gcl * ND + k0 + (e & 63)]);
    }
    // prefetch next k-chunk of w (one line per output column row)
    if (k0 + BK < ND) {
      const int gcl = min(colBase + tid, NC - 1);
      __builtin_prefetch(&w[(size_t)gcl * ND + k0 + BK], 0, 3);
    }
    // ---- convert to bf16 hi/lo and stage to LDS ----
#pragma unroll
    for (int j = 0; j < 2; ++j) {
      const int e  = (j * 256 + tid) * 4;
      const int r  = e >> 6;
      const int kk = e & 63;
      float xs[4] = {av[j].x, av[j].y, av[j].z, av[j].w};
#pragma unroll
      for (int i = 0; i < 4; ++i) store_hilo(&aHi[r][kk + i], &aLo[r][kk + i], xs[i]);
    }
#pragma unroll
    for (int it = 0; it < 16; ++it) {
      const int e  = (it * 256 + tid) * 4;
      const int c  = e >> 6;
      const int kk = e & 63;
      const bool valid = (colBase + c) < NC;
      float xs[4] = {bv[it].x, bv[it].y, bv[it].z, bv[it].w};
#pragma unroll
      for (int i = 0; i < 4; ++i)
        store_hilo(&bHi[c][kk + i], &bLo[c][kk + i], valid ? xs[i] : 0.f);
    }
    __syncthreads();

    // ---- 2 K-steps of 32; A frags reused across col tiles, B frags across row tiles ----
#pragma unroll
    for (int ks = 0; ks < 2; ++ks) {
      const int kb = ks * 32 + khalf * 8;
      v16bf Ah0 = ldfrag(&aHi[m][0],      kb);
      v16bf Al0 = ldfrag(&aLo[m][0],      kb);
      v16bf Ah1 = ldfrag(&aHi[16 + m][0], kb);
      v16bf Al1 = ldfrag(&aLo[16 + m][0], kb);
      {
        const int c = wave * 32 + m;
        v16bf Bh = ldfrag(&bHi[c][0], kb);
        v16bf Bl = ldfrag(&bLo[c][0], kb);
        acc00 = wmma3(acc00, Ah0, Al0, Bh, Bl);
        acc10 = wmma3(acc10, Ah1, Al1, Bh, Bl);
      }
      {
        const int c = wave * 32 + 16 + m;
        v16bf Bh = ldfrag(&bHi[c][0], kb);
        v16bf Bl = ldfrag(&bLo[c][0], kb);
        acc01 = wmma3(acc01, Ah0, Al0, Bh, Bl);
        acc11 = wmma3(acc11, Ah1, Al1, Bh, Bl);
      }
    }
    __syncthreads();
  }

  // ---- fused ArcFace epilogue ----
  // C/D layout: acc[r] -> row (r + khalf*8) of its 16-row tile, col (lane&15).
  float fnv0[8], fnv1[8];
#pragma unroll
  for (int r = 0; r < 8; ++r) {
    fnv0[r] = fn[rowBase + khalf * 8 + r];
    fnv1[r] = fn[rowBase + 16 + khalf * 8 + r];
  }

  const int gc0 = colBase + wave * 32 + m;
  const int gc1 = gc0 + 16;
  const bool v0 = gc0 < NC;
  const bool v1 = gc1 < NC;
  const float wn0 = v0 ? wn[gc0] : 1.f;
  const float wn1 = v1 ? wn[gc1] : 1.f;

  float out00[8], out01[8], out10[8], out11[8];
#pragma unroll
  for (int r = 0; r < 8; ++r) {
    const int b0 = rowBase + khalf * 8 + r;
    const int b1 = b0 + 16;
    const float l00 = v0 ? labels[(size_t)b0 * NC + gc0] : 0.f;
    const float l01 = v1 ? labels[(size_t)b0 * NC + gc1] : 0.f;
    const float l10 = v0 ? labels[(size_t)b1 * NC + gc0] : 0.f;
    const float l11 = v1 ? labels[(size_t)b1 * NC + gc1] : 0.f;
    out00[r] = arcface_out(acc00[r], fnv0[r], wn0, l00, v0);
    out01[r] = arcface_out(acc01[r], fnv0[r], wn1, l01, v1);
    out10[r] = arcface_out(acc10[r], fnv1[r], wn0, l10, v0);
    out11[r] = arcface_out(acc11[r], fnv1[r], wn1, l11, v1);
  }

  // ---- block-level row max over 256 cols ----
  float mx0[8], mx1[8];
#pragma unroll
  for (int r = 0; r < 8; ++r) {
    mx0[r] = fmaxf(out00[r], out01[r]);
    mx1[r] = fmaxf(out10[r], out11[r]);
  }
#pragma unroll
  for (int d = 1; d < 16; d <<= 1)
#pragma unroll
    for (int r = 0; r < 8; ++r) {
      mx0[r] = fmaxf(mx0[r], __shfl_xor(mx0[r], d, 32));
      mx1[r] = fmaxf(mx1[r], __shfl_xor(mx1[r], d, 32));
    }
  if ((lane & 15) == 0) {
#pragma unroll
    for (int r = 0; r < 8; ++r) {
      sRed[wave][khalf * 8 + r]      = mx0[r];
      sRed[wave][16 + khalf * 8 + r] = mx1[r];
    }
  }
  __syncthreads();
  if (tid < BM) {
    float M = -INFINITY;
#pragma unroll
    for (int wv = 0; wv < 8; ++wv) M = fmaxf(M, sRed[wv][tid]);
    rowMax[tid] = M;
  }
  __syncthreads();

  // ---- sum of exp(x - rowMax) ----
  float sm0[8], sm1[8];
#pragma unroll
  for (int r = 0; r < 8; ++r) {
    const float RM0 = rowMax[khalf * 8 + r];
    const float RM1 = rowMax[16 + khalf * 8 + r];
    sm0[r] = expf(out00[r] - RM0) + expf(out01[r] - RM0);  // exp(-inf - RM) = 0
    sm1[r] = expf(out10[r] - RM1) + expf(out11[r] - RM1);
  }
#pragma unroll
  for (int d = 1; d < 16; d <<= 1)
#pragma unroll
    for (int r = 0; r < 8; ++r) {
      sm0[r] += __shfl_xor(sm0[r], d, 32);
      sm1[r] += __shfl_xor(sm1[r], d, 32);
    }
  __syncthreads();  // sRed reuse
  if ((lane & 15) == 0) {
#pragma unroll
    for (int r = 0; r < 8; ++r) {
      sRed[wave][khalf * 8 + r]      = sm0[r];
      sRed[wave][16 + khalf * 8 + r] = sm1[r];
    }
  }
  __syncthreads();
  if (tid < BM) {
    float S = 0.f;
#pragma unroll
    for (int wv = 0; wv < 8; ++wv) S += sRed[wv][tid];
    const size_t o = ((size_t)(rowBase + tid) * NCB + blockIdx.y) * 2;
    stats[o]     = rowMax[tid];
    stats[o + 1] = S;
  }
}

__global__ void init_out_kernel(float* out) { out[0] = 0.f; }

// ---------------------------------------------------------------------------
// Per-row log-sum-exp merge + exact fp32 recompute of <=8 gathered logits.
// One wave per row.
// ---------------------------------------------------------------------------
__global__ __launch_bounds__(256) void finalize_kernel(
    const float* __restrict__ f, const float* __restrict__ w,
    const float* __restrict__ fn, const float* __restrict__ wn,
    const int* __restrict__ pinds, const int* __restrict__ lengths,
    const float* __restrict__ stats, float* __restrict__ out) {
  const int lane = threadIdx.x & 31;
  const int wave = threadIdx.x >> 5;
  const int b = blockIdx.x * 8 + wave;

  float M = -INFINITY;
  for (int i = lane; i < NCB; i += 32) M = fmaxf(M, stats[((size_t)b * NCB + i) * 2]);
#pragma unroll
  for (int d = 1; d < 32; d <<= 1) M = fmaxf(M, __shfl_xor(M, d, 32));
  float S = 0.f;
  for (int i = lane; i < NCB; i += 32) {
    const size_t o = ((size_t)b * NCB + i) * 2;
    S += stats[o + 1] * expf(stats[o] - M);
  }
#pragma unroll
  for (int d = 1; d < 32; d <<= 1) S += __shfl_xor(S, d, 32);
  const float logZ = M + logf(S);

  const int len = lengths[b];
  const float fnb = fn[b];
  float accv = 0.f;
  for (int j = 0; j < LMAXC; ++j) {
    if (j < len) {
      const int c = pinds[(size_t)b * LMAXC + j];
      float d = 0.f;
#pragma unroll
      for (int kk = 0; kk < 16; ++kk) {
        const int k = lane + kk * 32;
        d += f[(size_t)b * ND + k] * w[(size_t)c * ND + k];
      }
#pragma unroll
      for (int dd = 1; dd < 32; dd <<= 1) d += __shfl_xor(d, dd, 32);
      float cosv = d / fmaxf(fnb * wn[c], 1e-8f);
      float sine = sqrtf(fminf(fmaxf(1.f - cosv * cosv, 0.f), 1.f));
      float phi  = cosv * COS_M - sine * SIN_M;
      phi = (cosv > THR) ? phi : (cosv - MMV);
      accv += (S_SCALE * phi - logZ);  // gathered entries carry label==1
    }
  }
  if (lane == 0) {
    const float L = (float)len;
    atomicAdd(out, (-accv / (L * L)) / (float)NB);
  }
}

extern "C" void kernel_launch(void* const* d_in, const int* in_sizes, int n_in,
                              void* d_out, int out_size, void* d_ws, size_t ws_size,
                              hipStream_t stream) {
  (void)in_sizes; (void)n_in; (void)out_size; (void)ws_size;
  const float* f       = (const float*)d_in[0];
  const float* labels  = (const float*)d_in[1];
  const float* w       = (const float*)d_in[2];
  const int*   pinds   = (const int*)d_in[3];
  const int*   lengths = (const int*)d_in[4];
  float* out = (float*)d_out;

  float* ws    = (float*)d_ws;
  float* wn    = ws;                 // NC floats
  float* fn    = ws + NC;            // NB floats
  float* stats = ws + NC + NB;       // NB * NCB * 2 floats (~1.3 MB)

  norms_kernel<<<(NC + NB + 7) / 8, 256, 0, stream>>>(f, w, wn, fn);
  dim3 grid(NB / BM, NCB);
  gemm_stats_kernel<<<grid, 256, 0, stream>>>(f, labels, w, fn, wn, stats);
  init_out_kernel<<<1, 1, 0, stream>>>(out);
  finalize_kernel<<<NB / 8, 256, 0, stream>>>(f, w, fn, wn, pinds, lengths, stats, out);
}
